// RelativePositionMultiHeadAttention_2473901162543
// MI455X (gfx1250) — matrix operand order repaired and checked by reading, not compiled, not run on hardware
//
#include <hip/hip_runtime.h>

#define CH_   1024
#define T_    1024
#define B_    4
#define H_    16
#define KC_   64

typedef __attribute__((ext_vector_type(16))) __bf16 v16bf;
typedef __attribute__((ext_vector_type(8)))  __bf16 v8bf;
typedef __attribute__((ext_vector_type(8)))  float  v8f;
typedef __attribute__((ext_vector_type(4)))  unsigned v4u;
typedef __attribute__((ext_vector_type(8)))  int      v8i;
typedef __attribute__((ext_vector_type(4)))  int      v4i;

#if defined(__HIP_DEVICE_COMPILE__) && __has_builtin(__builtin_amdgcn_tensor_load_to_lds) && __has_builtin(__builtin_amdgcn_s_wait_tensorcnt)
#define HAVE_TDM 1
#else
#define HAVE_TDM 0
#endif

__device__ __forceinline__ v16bf ld2x8(const __bf16* p0, const __bf16* p1) {
  v16bf r;
  ((v8bf*)&r)[0] = *(const v8bf*)p0;
  ((v8bf*)&r)[1] = *(const v8bf*)p1;
  return r;
}

__device__ __forceinline__ v8f wmma_bf16(v16bf a, v16bf b, v8f c) {
  return __builtin_amdgcn_wmma_f32_16x16x32_bf16(false, a, false, b,
                                                 (short)0, c, false, false);
}

#if HAVE_TDM
// Build a D# (ISA 08_async_tensor §8) and issue TENSOR_LOAD_TO_LDS.
// 2-byte elements; 1-D tile when tile_d1==0, else 2-D tile with row stride
// stride0 (in elements). Tensor dims set large (tiles never clipped).
__device__ __forceinline__ void tdm_load(unsigned lds_byte_addr, const void* gptr,
                                         unsigned tile_d0, unsigned tile_d1,
                                         unsigned long long stride0) {
  unsigned long long ga = (unsigned long long)(size_t)gptr;
  const unsigned td0 = 1u << 20, td1 = 1u << 20;
  v4u g0;
  g0[0] = 1u;                                           // count=1, user mode
  g0[1] = lds_byte_addr;                                // lds_addr
  g0[2] = (unsigned)ga;                                 // global_addr[31:0]
  g0[3] = (unsigned)((ga >> 32) & 0x1FFFFFFu)           // global_addr[56:32]
          | 0x80000000u;                                // type=2 ("image")
  v8i g1;
  g1[0] = (int)(1u << 16);                              // data_size=1 (2 bytes)
  g1[1] = (int)((td0 & 0xFFFFu) << 16);                 // tensor_dim0[15:0]
  g1[2] = (int)(((td0 >> 16) & 0xFFFFu) | ((td1 & 0xFFFFu) << 16));
  g1[3] = (int)(((td1 >> 16) & 0xFFFFu) | (tile_d0 << 16));  // tile_dim0
  g1[4] = (int)(tile_d1 & 0xFFFFu);                     // tile_dim1, tile_dim2=0
  g1[5] = (int)(unsigned)(stride0 & 0xFFFFFFFFull);     // tensor_dim0_stride
  g1[6] = (int)(unsigned)((stride0 >> 32) & 0xFFFFull); // stride0 hi, stride1=0
  g1[7] = 0;
  v4i z = {0, 0, 0, 0};
#if __clang_major__ >= 23
  v8i z8 = {0, 0, 0, 0, 0, 0, 0, 0};
  __builtin_amdgcn_tensor_load_to_lds(g0, g1, z, z, z8, 0);
#else
  __builtin_amdgcn_tensor_load_to_lds(g0, g1, z, z, 0);
#endif
}
#define WAIT_TENSORCNT(n) __builtin_amdgcn_s_wait_tensorcnt((short)(n))
#endif

// ---------------------------------------------------------------- pack f32 -> bf16
__global__ __launch_bounds__(256) void pack_w(const float* __restrict__ src,
                                              __bf16* __restrict__ dst, int n) {
  int i = blockIdx.x * 256 + threadIdx.x;
  if (i < n) dst[i] = (__bf16)src[i];
}

// src [B][CH][T] f32 -> dst [B][T][CH] bf16 (channels contiguous for B-fragments)
__global__ __launch_bounds__(256) void pack_xT(const float* __restrict__ src,
                                               __bf16* __restrict__ dst) {
  int i = blockIdx.x * 256 + threadIdx.x;
  int t  = i & (T_ - 1);
  int ch = (i >> 10) & (CH_ - 1);
  int b  = i >> 20;
  dst[((size_t)(b * T_ + t)) * CH_ + ch] = (__bf16)src[i];
}

// ---------------------------------------------------------------- WMMA GEMM (2x2 tiles)
// D[m, col] = sum_k W[m,k] * X[col, k] + bias[m]
// Each wave computes a 32x32 tile (2x2 WMMA accumulators) to double the
// FLOP/byte vs single-tile waves.
// mode 0: bf16 out [B,H,T,KC]   (q/k)
// mode 1: bf16 out [B,CH,T]     (v, pre-transposed for P*V)
// mode 2: f32  out [B,CH,T]     (final projection)
__global__ __launch_bounds__(256) void gemm_wmma(const __bf16* __restrict__ W,
                                                 const __bf16* __restrict__ X,
                                                 const float* __restrict__ bias,
                                                 void* __restrict__ out, int mode) {
  const int lane = threadIdx.x & 31;
  const int wave = threadIdx.x >> 5;
  const int half = lane >> 4;
  const int n    = lane & 15;
  const int gw = blockIdx.x * 8 + wave;     // 512 blocks * 8 waves = 32*128 super-tiles
  const int mt = gw & 31;
  const int nt = gw >> 5;
  const int m0 = mt * 32;
  const int col0 = nt * 32;

  const __bf16* ar0 = W + (size_t)(m0 + n) * CH_;
  const __bf16* ar1 = W + (size_t)(m0 + 16 + n) * CH_;
  const __bf16* br0 = X + (size_t)(col0 + n) * CH_;
  const __bf16* br1 = X + (size_t)(col0 + 16 + n) * CH_;

  v8f acc[2][2] = {{{}, {}}, {{}, {}}};
  for (int k0 = 0; k0 < CH_; k0 += 32) {
    v16bf a0 = ld2x8(ar0 + k0 + half * 8, ar0 + k0 + 16 + half * 8);
    v16bf a1 = ld2x8(ar1 + k0 + half * 8, ar1 + k0 + 16 + half * 8);
    v16bf b0 = ld2x8(br0 + k0 + half * 8, br0 + k0 + 16 + half * 8);
    v16bf b1 = ld2x8(br1 + k0 + half * 8, br1 + k0 + 16 + half * 8);
    acc[0][0] = wmma_bf16(a0, b0, acc[0][0]);
    acc[0][1] = wmma_bf16(a0, b1, acc[0][1]);
    acc[1][0] = wmma_bf16(a1, b0, acc[1][0]);
    acc[1][1] = wmma_bf16(a1, b1, acc[1][1]);
  }

#pragma unroll
  for (int mi = 0; mi < 2; ++mi)
#pragma unroll
    for (int ni = 0; ni < 2; ++ni) {
      const int oc0 = m0 + 16 * mi + half * 8;
      const int col = col0 + 16 * ni + n;
      const int b = col >> 10;
      const int t = col & (T_ - 1);
#pragma unroll
      for (int r = 0; r < 8; ++r) acc[mi][ni][r] += bias[oc0 + r];
      if (mode == 0) {
        __bf16* o = (__bf16*)out;
        int h = oc0 >> 6, d0 = oc0 & 63;
        v8bf pk;
#pragma unroll
        for (int r = 0; r < 8; ++r) pk[r] = (__bf16)acc[mi][ni][r];
        *(v8bf*)&o[(((size_t)(b * H_ + h) * T_ + t) * KC_) + d0] = pk;
      } else if (mode == 1) {
        __bf16* o = (__bf16*)out;
#pragma unroll
        for (int r = 0; r < 8; ++r)
          o[((size_t)(b * CH_ + oc0 + r)) * T_ + t] = (__bf16)acc[mi][ni][r];
      } else {
        float* o = (float*)out;
#pragma unroll
        for (int r = 0; r < 8; ++r)
          o[((size_t)(b * CH_ + oc0 + r)) * T_ + t] = acc[mi][ni][r];
      }
    }
}

// ---------------------------------------------------------------- rel-K bias band
// rb[bh, t, j] = scale * sum_d q[bh,t,d] * relk[j,d]   (j = off+4, off in [-4,4])
__global__ __launch_bounds__(256) void rel_bias(const __bf16* __restrict__ q,
                                                const float* __restrict__ relk,
                                                float* __restrict__ rb) {
  int i = blockIdx.x * 256 + threadIdx.x;     // over B*H*T
  const __bf16* qr = q + (size_t)i * KC_;
  float qv[KC_];
#pragma unroll
  for (int dv = 0; dv < 8; ++dv) {
    v8bf v = ((const v8bf*)qr)[dv];
#pragma unroll
    for (int e = 0; e < 8; ++e) qv[dv * 8 + e] = (float)v[e];
  }
  const float scale = 0.125f;                 // 1/sqrt(64)
#pragma unroll
  for (int j = 0; j < 9; ++j) {
    float s = 0.f;
#pragma unroll
    for (int d = 0; d < KC_; ++d) s += qv[d] * relk[j * KC_ + d];
    rb[(size_t)i * 9 + j] = s * scale;
  }
}

// ---------------------------------------------------------------- fused flash attention
// grid: B*H*(T/128) = 512 blocks, 256 threads (8 waves, 16 queries each).
// K/V tiles are staged block-wide into LDS by the Tensor Data Mover
// (double-buffered, overlapping DMA with WMMA), so the 8 waves share one copy.
__global__ __launch_bounds__(256) void attention(const __bf16* __restrict__ Q,   // [B,H,T,KC]
                                                 const __bf16* __restrict__ Kb,  // [B,H,T,KC]
                                                 const __bf16* __restrict__ Vt,  // [B,CH,T] == [B,H,KC,T]
                                                 const float* __restrict__ rb,   // [B,H,T,9]
                                                 const float* __restrict__ relv, // [9,KC]
                                                 __bf16* __restrict__ att) {     // [B,T,CH]
  __shared__ __align__(16) __bf16 sK[2][32][64];   // staged K tiles (4KB each)
  __shared__ __align__(16) __bf16 sV[2][64][32];   // staged V tiles (4KB each)
  __shared__ __align__(16) __bf16 sP[8][16][32];   // per-wave P staging (D->A relayout)
  __shared__ float sStat[8][16][2];                // per-row (max, sum)
  __shared__ float sPb[8][16][9];                  // banded probabilities

  const int lane = threadIdx.x & 31;
  const int wave = threadIdx.x >> 5;
  const int half = lane >> 4;
  const int n    = lane & 15;
  const int bh    = blockIdx.x >> 3;
  const int ttile = blockIdx.x & 7;
  const int t0 = ttile * 128 + wave * 16;
  const int b = bh >> 4, h = bh & 15;
  const float scale = 0.125f;

  const __bf16* qb  = Q  + (size_t)bh * T_ * KC_;
  const __bf16* kb  = Kb + (size_t)bh * T_ * KC_;
  const __bf16* vb  = Vt + ((size_t)b * CH_ + h * KC_) * T_;
  const float*  rbb = rb + (size_t)bh * T_ * 9;

  // Q A-fragments (16 queries x K=64), resident for the whole pass
  const __bf16* qrow = qb + (size_t)(t0 + n) * KC_;
  v16bf aq0 = ld2x8(qrow + half * 8,      qrow + 16 + half * 8);
  v16bf aq1 = ld2x8(qrow + 32 + half * 8, qrow + 48 + half * 8);

  float rowm[8], rowl[8];
#pragma unroll
  for (int r = 0; r < 8; ++r) { rowm[r] = -__builtin_inff(); rowl[r] = 0.f; }
  v8f O[4] = {{}, {}, {}, {}};

  int cur = 0;
#if HAVE_TDM
  if (wave == 0) {
    tdm_load((unsigned)(size_t)&sK[0][0][0], kb, 2048, 0, 1u << 20);      // 1-D 4KB
    tdm_load((unsigned)(size_t)&sV[0][0][0], vb, 32, 64, (unsigned long long)T_);
  }
#endif

  for (int it = 0; it < T_ / 32; ++it) {
    const int s0 = it * 32;
    const int nxt = cur ^ 1;
#if HAVE_TDM
    if (wave == 0) {
      if (it + 1 < T_ / 32) {
        tdm_load((unsigned)(size_t)&sK[nxt][0][0], kb + (size_t)(s0 + 32) * KC_,
                 2048, 0, 1u << 20);
        tdm_load((unsigned)(size_t)&sV[nxt][0][0], vb + (s0 + 32),
                 32, 64, (unsigned long long)T_);
        WAIT_TENSORCNT(2);          // current tile's 2 DMAs complete
      } else {
        WAIT_TENSORCNT(0);
      }
    }
#else
    {   // fallback: cooperative staging through VGPRs
      const int tid = threadIdx.x;
      ((v8bf*)&sK[cur][0][0])[tid] = ((const v8bf*)(kb + (size_t)s0 * KC_))[tid];
      const int row = tid >> 2, ch = tid & 3;
      *(v8bf*)&sV[cur][row][ch * 8] =
          *(const v8bf*)(vb + (size_t)row * T_ + s0 + ch * 8);
    }
#endif
    __syncthreads();

    // ---- scores: two 16x16 tiles, K=64 chained, B-fragments from LDS
    const __bf16* kr0 = &sK[cur][n][0];
    const __bf16* kr1 = &sK[cur][16 + n][0];
    v8f Sa = {}, Sb = {};
    Sa = wmma_bf16(aq0, ld2x8(kr0 + half * 8, kr0 + 16 + half * 8), Sa);
    Sa = wmma_bf16(aq1, ld2x8(kr0 + 32 + half * 8, kr0 + 48 + half * 8), Sa);
    Sb = wmma_bf16(aq0, ld2x8(kr1 + half * 8, kr1 + 16 + half * 8), Sb);
    Sb = wmma_bf16(aq1, ld2x8(kr1 + 32 + half * 8, kr1 + 48 + half * 8), Sb);
#pragma unroll
    for (int r = 0; r < 8; ++r) { Sa[r] *= scale; Sb[r] *= scale; }

    // ---- relative-K bias: only the |s-t|<=4 band is nonzero
    if (s0 + 31 >= t0 - 4 && s0 <= t0 + 19) {
#pragma unroll
      for (int r = 0; r < 8; ++r) {
        int tq = t0 + half * 8 + r;
        int off = (s0 + n) - tq;
        if (off >= -4 && off <= 4) Sa[r] += rbb[tq * 9 + off + 4];
        off = (s0 + 16 + n) - tq;
        if (off >= -4 && off <= 4) Sb[r] += rbb[tq * 9 + off + 4];
      }
    }

    // ---- online softmax (rows live in 16-lane halves)
#pragma unroll
    for (int r = 0; r < 8; ++r) {
      float tm = fmaxf(Sa[r], Sb[r]);
#pragma unroll
      for (int m = 1; m < 16; m <<= 1) tm = fmaxf(tm, __shfl_xor(tm, m, 32));
      float mnew = fmaxf(rowm[r], tm);
      float pa = __expf(Sa[r] - mnew);
      float pb = __expf(Sb[r] - mnew);
      float rs = pa + pb;
#pragma unroll
      for (int m = 1; m < 16; m <<= 1) rs += __shfl_xor(rs, m, 32);
      float alpha = __expf(rowm[r] - mnew);
      rowl[r] = rowl[r] * alpha + rs;
      rowm[r] = mnew;
      O[0][r] *= alpha; O[1][r] *= alpha; O[2][r] *= alpha; O[3][r] *= alpha;
      int M = half * 8 + r;
      sP[wave][M][n] = (__bf16)pa;
      sP[wave][M][16 + n] = (__bf16)pb;
    }

    // ---- P*V: D-layout -> A-layout via per-wave LDS (DS ops in-order per wave)
    const __bf16* prow = &sP[wave][n][0];
    v16bf ap = ld2x8(prow + half * 8, prow + 16 + half * 8);
#pragma unroll
    for (int j = 0; j < 4; ++j) {
      const __bf16* vr = &sV[cur][16 * j + n][half * 8];
      O[j] = wmma_bf16(ap, ld2x8(vr, vr + 16), O[j]);
    }
    __syncthreads();          // tile fully consumed before buffer reuse
    cur = nxt;
  }

  // ---- publish per-row stats
  if (n == 0) {
#pragma unroll
    for (int r = 0; r < 8; ++r) {
      sStat[wave][half * 8 + r][0] = rowm[r];
      sStat[wave][half * 8 + r][1] = rowl[r];
    }
  }

  // ---- banded rel-V probabilities: p_attn[t, t+off], off in [-4,4]
  {
    int row = n, tq = t0 + row;
    int jlo = half ? 5 : 0, jhi = half ? 9 : 5;
    v8bf qv8[8];
    const __bf16* q1 = qb + (size_t)tq * KC_;
#pragma unroll
    for (int dv = 0; dv < 8; ++dv) qv8[dv] = ((const v8bf*)q1)[dv];
    for (int j = jlo; j < jhi; ++j) {
      int s = tq + j - 4;
      float pv = 0.f;
      if (s >= 0 && s < T_) {
        const __bf16* k1 = kb + (size_t)s * KC_;
        float acc = 0.f;
#pragma unroll
        for (int dv = 0; dv < 8; ++dv) {
          v8bf kv = ((const v8bf*)k1)[dv];
#pragma unroll
          for (int e = 0; e < 8; ++e) acc += (float)qv8[dv][e] * (float)kv[e];
        }
        float score = acc * scale + rbb[tq * 9 + j];
        pv = __expf(score - sStat[wave][row][0]) / sStat[wave][row][1];
      }
      sPb[wave][row][j] = pv;
    }
  }

  // ---- finalize: O/l + band * rel_v, store bf16 [B,T,CH] for Wo GEMM
  float inv[8];
#pragma unroll
  for (int r = 0; r < 8; ++r) inv[r] = 1.f / rowl[r];
#pragma unroll
  for (int j = 0; j < 4; ++j) {
    int d = 16 * j + n;
    float rv[9];
#pragma unroll
    for (int jj = 0; jj < 9; ++jj) rv[jj] = relv[jj * KC_ + d];
#pragma unroll
    for (int r = 0; r < 8; ++r) {
      float o = O[j][r] * inv[r];
#pragma unroll
      for (int jj = 0; jj < 9; ++jj) o += sPb[wave][half * 8 + r][jj] * rv[jj];
      int tq = t0 + half * 8 + r;
      att[((size_t)(b * T_ + tq)) * CH_ + h * KC_ + d] = (__bf16)o;
    }
  }
}

// ---------------------------------------------------------------- launch
extern "C" void kernel_launch(void* const* d_in, const int* in_sizes, int n_in,
                              void* d_out, int out_size, void* d_ws, size_t ws_size,
                              hipStream_t stream) {
  (void)in_sizes; (void)n_in; (void)out_size; (void)ws_size;
  const float* x    = (const float*)d_in[0];
  const float* c    = (const float*)d_in[1];
  const float* Wq   = (const float*)d_in[2];
  const float* bq   = (const float*)d_in[3];
  const float* Wk   = (const float*)d_in[4];
  const float* bk   = (const float*)d_in[5];
  const float* Wv   = (const float*)d_in[6];
  const float* bv   = (const float*)d_in[7];
  const float* Wo   = (const float*)d_in[8];
  const float* bo   = (const float*)d_in[9];
  const float* relk = (const float*)d_in[10];
  const float* relv = (const float*)d_in[11];

  char* ws = (char*)d_ws;
  size_t off = 0;
  auto alloc = [&](size_t bytes) {
    void* p = ws + off;
    off += (bytes + 255) & ~(size_t)255;
    return p;
  };
  const size_t WBYTES = (size_t)CH_ * CH_ * 2;
  const size_t XBYTES = (size_t)B_ * T_ * CH_ * 2;
  __bf16* Wq_bf = (__bf16*)alloc(WBYTES);
  __bf16* Wk_bf = (__bf16*)alloc(WBYTES);
  __bf16* Wv_bf = (__bf16*)alloc(WBYTES);
  __bf16* Wo_bf = (__bf16*)alloc(WBYTES);
  __bf16* xT_bf = (__bf16*)alloc(XBYTES);
  __bf16* cT_bf = (__bf16*)alloc(XBYTES);
  __bf16* q_bf  = (__bf16*)alloc(XBYTES);    // [B,H,T,KC]
  __bf16* k_bf  = (__bf16*)alloc(XBYTES);    // [B,H,T,KC]
  __bf16* vT_bf = (__bf16*)alloc(XBYTES);    // [B,CH,T]
  __bf16* att_bf = (__bf16*)alloc(XBYTES);   // [B,T,CH]
  float*  rb    = (float*)alloc((size_t)B_ * H_ * T_ * 9 * 4);

  const int WN = CH_ * CH_;
  pack_w<<<WN / 256, 256, 0, stream>>>(Wq, Wq_bf, WN);
  pack_w<<<WN / 256, 256, 0, stream>>>(Wk, Wk_bf, WN);
  pack_w<<<WN / 256, 256, 0, stream>>>(Wv, Wv_bf, WN);
  pack_w<<<WN / 256, 256, 0, stream>>>(Wo, Wo_bf, WN);
  pack_xT<<<(B_ * CH_ * T_) / 256, 256, 0, stream>>>(x, xT_bf);
  pack_xT<<<(B_ * CH_ * T_) / 256, 256, 0, stream>>>(c, cT_bf);

  // Q, K, V projections (WMMA, 32x32 per wave)
  gemm_wmma<<<512, 256, 0, stream>>>(Wq_bf, xT_bf, bq, (void*)q_bf, 0);
  gemm_wmma<<<512, 256, 0, stream>>>(Wk_bf, cT_bf, bk, (void*)k_bf, 0);
  gemm_wmma<<<512, 256, 0, stream>>>(Wv_bf, cT_bf, bv, (void*)vT_bf, 1);

  // rel-K band bias
  rel_bias<<<(B_ * H_ * T_) / 256, 256, 0, stream>>>(q_bf, relk, rb);

  // fused flash attention with TDM-staged K/V and relative-position band terms
  attention<<<B_ * H_ * (T_ / 128), 256, 0, stream>>>(q_bf, k_bf, vT_bf, rb, relv, att_bf);

  // output projection (WMMA, f32 out)
  gemm_wmma<<<512, 256, 0, stream>>>(Wo_bf, att_bf, bo, d_out, 2);
}
